// AdvancedSpikingChatModel_33071248179959
// MI455X (gfx1250) — matrix operand order repaired and checked by reading, not compiled, not run on hardware
//
#include <hip/hip_runtime.h>
#include <hip/hip_bf16.h>
#include <math.h>

// ---------------------------------------------------------------------------
// AdvancedSpikingChatModel on gfx1250 (MI455X).
// All matmuls via v_wmma_f32_16x16x32_f16 (wave32 WMMA). LIF scans are
// elementwise over T=16. Head GEMM + head LIF fused (logits never hit HBM).
// GEMM: 128x64 block tile, 32x32 wave tile (4 WMMA/K-step, 2x fragment
// reuse), double-buffered LDS to overlap global loads with matrix math.
// ---------------------------------------------------------------------------

typedef _Float16 h16;
typedef __attribute__((ext_vector_type(16))) _Float16 v16h;
typedef __attribute__((ext_vector_type(8)))  _Float16 v8h;
typedef __attribute__((ext_vector_type(8)))  float    v8f;

#define B_  2
#define S_  128
#define D_  256
#define T_  16
#define L_  6
#define F_  1024
#define V_  32000
#define N_  (B_*S_)    // 256 rows (b*S+s)
#define NT_ (N_*T_)    // 4096 GEMM rows

#define WMMA16(a,b,c) \
  __builtin_amdgcn_wmma_f32_16x16x32_f16(false,(a),false,(b),(short)0,(c),false,false)

union FragH { v16h v; v8h h[2]; };

// ---------------------------------------------------------------------------
// Weight convert + transpose: in (K x N) f32 row-major -> out (N x K) f16
// ---------------------------------------------------------------------------
__global__ void convT_kernel(const float* __restrict__ in, h16* __restrict__ out,
                             int K, int N) {
  int i = blockIdx.x * 256 + threadIdx.x;
  if (i < K * N) {
    int n = i / K, k = i - n * K;
    out[i] = (h16)in[(size_t)k * N + n];
  }
}

// ---------------------------------------------------------------------------
// Spike encoder: rate + temporal coding -> x (N, T, D) as f32 and f16
// ---------------------------------------------------------------------------
__global__ void encode_kernel(const int* __restrict__ ids,
                              const float* __restrict__ tok,
                              const float* __restrict__ pos,
                              const float* __restrict__ noise,
                              const float* __restrict__ unif,
                              float* __restrict__ XF, h16* __restrict__ XH) {
  int i = blockIdx.x * 256 + threadIdx.x;          // over N_*D_
  if (i >= N_ * D_) return;
  int d = i & (D_ - 1);
  int n = i >> 8;                                   // b*S+s
  int s = n & (S_ - 1);
  float e  = tok[(size_t)ids[n] * D_ + d] + pos[s * D_ + d];
  float sg = 1.0f / (1.0f + expf(-e));
  float p  = sg * 0.8f + 0.1f + noise[i] * 0.05f;
  p = fminf(fmaxf(p, 0.0f), 1.0f);
  int ti = (int)(sg * (float)(T_ - 1));
#pragma unroll
  for (int t = 0; t < T_; ++t) {
    float rate = (unif[(size_t)t * (N_ * D_) + i] < p) ? 1.0f : 0.0f;
    float temp = (t == ti) ? 1.0f : 0.0f;
    float sp   = (0.7f * rate + 0.3f * temp) > 0.5f ? 1.0f : 0.0f;
    size_t a = ((size_t)n * T_ + t) * D_ + d;
    XF[a] = sp;
    XH[a] = (h16)sp;
  }
}

// ---------------------------------------------------------------------------
// WMMA GEMM: C(M,N) = A(M,K) @ Bt(N,K)^T (+bias[n]).  f16 in, f32 out.
// Block tile 128x64, 8 waves in 4x2; wave tile 32x32 = 4 WMMA per K-step.
// Double-buffered LDS (24 KB) overlaps next-tile loads with WMMA.
// ---------------------------------------------------------------------------
#define KC 32
__global__ void __launch_bounds__(256)
gemm_wmma(const h16* __restrict__ A, const h16* __restrict__ Bt,
          const float* __restrict__ bias, float* __restrict__ C,
          int M, int N, int K) {
  __shared__ __align__(16) h16 sA[2][128 * KC];   // 2 x 8 KB
  __shared__ __align__(16) h16 sB[2][64 * KC];    // 2 x 4 KB
  const int m0 = blockIdx.x * 128, n0 = blockIdx.y * 64;
  const int tid  = threadIdx.x;
  const int w    = tid >> 5, lane = tid & 31;
  const int wm   = w >> 1,  wn   = w & 1;        // wave rows wm*32, cols wn*32
  const int mrow = lane & 15, hh = lane >> 4;    // ISA A/B fragment mapping
  const int ldr  = tid >> 2, ldc = (tid & 3) * 8;

  v8f c00 = {}, c01 = {}, c10 = {}, c11 = {};

  // Preload K-tile 0
  *(v8h*)&sA[0][ldr * KC + ldc]        = *(const v8h*)&A[(size_t)(m0 + ldr) * K + ldc];
  *(v8h*)&sA[0][(ldr + 64) * KC + ldc] = *(const v8h*)&A[(size_t)(m0 + ldr + 64) * K + ldc];
  *(v8h*)&sB[0][ldr * KC + ldc]        = *(const v8h*)&Bt[(size_t)(n0 + ldr) * K + ldc];
  __syncthreads();

  for (int k0 = 0; k0 < K; k0 += KC) {
    const int buf = (k0 / KC) & 1, nxt = buf ^ 1;
    if (k0 + KC < K) {  // issue next tile's global loads before computing
      *(v8h*)&sA[nxt][ldr * KC + ldc] =
          *(const v8h*)&A[(size_t)(m0 + ldr) * K + k0 + KC + ldc];
      *(v8h*)&sA[nxt][(ldr + 64) * KC + ldc] =
          *(const v8h*)&A[(size_t)(m0 + ldr + 64) * K + k0 + KC + ldc];
      *(v8h*)&sB[nxt][ldr * KC + ldc] =
          *(const v8h*)&Bt[(size_t)(n0 + ldr) * K + k0 + KC + ldc];
      if (k0 + 2 * KC < K)   // CDNA5 global_prefetch for the tile after next
        __builtin_prefetch(&A[(size_t)(m0 + ldr) * K + k0 + 2 * KC + ldc], 0, 3);
    }
    FragH a0, a1, b0, b1;
    // A 16x32 f16: lane holds row mrow; K chunks [hh*8,+8) and [16+hh*8,+8).
    a0.h[0] = *(const v8h*)&sA[buf][(wm * 32 + mrow) * KC + hh * 8];
    a0.h[1] = *(const v8h*)&sA[buf][(wm * 32 + mrow) * KC + 16 + hh * 8];
    a1.h[0] = *(const v8h*)&sA[buf][(wm * 32 + 16 + mrow) * KC + hh * 8];
    a1.h[1] = *(const v8h*)&sA[buf][(wm * 32 + 16 + mrow) * KC + 16 + hh * 8];
    // B 32x16 f16: lane holds col mrow; K chunks [hh*8,+8) and [16+hh*8,+8).
    b0.h[0] = *(const v8h*)&sB[buf][(wn * 32 + mrow) * KC + hh * 8];
    b0.h[1] = *(const v8h*)&sB[buf][(wn * 32 + mrow) * KC + 16 + hh * 8];
    b1.h[0] = *(const v8h*)&sB[buf][(wn * 32 + 16 + mrow) * KC + hh * 8];
    b1.h[1] = *(const v8h*)&sB[buf][(wn * 32 + 16 + mrow) * KC + 16 + hh * 8];
    c00 = WMMA16(a0.v, b0.v, c00);
    c01 = WMMA16(a0.v, b1.v, c01);
    c10 = WMMA16(a1.v, b0.v, c10);
    c11 = WMMA16(a1.v, b1.v, c11);
    __syncthreads();
  }

  const int col0 = n0 + wn * 32 + mrow, col1 = col0 + 16;
  const float bi0 = bias ? bias[col0] : 0.0f;
  const float bi1 = bias ? bias[col1] : 0.0f;
#pragma unroll
  for (int r = 0; r < 8; ++r) {
    int mA = m0 + wm * 32 + r + 8 * hh;           // ISA C layout
    int mB = mA + 16;
    C[(size_t)mA * N + col0] = c00[r] + bi0;
    C[(size_t)mA * N + col1] = c01[r] + bi1;
    C[(size_t)mB * N + col0] = c10[r] + bi0;
    C[(size_t)mB * N + col1] = c11[r] + bi1;
  }
}

// ---------------------------------------------------------------------------
// Attention recurrence: LIF on R/K/V streams + h decay, emit rh = r*h (f16)
// ---------------------------------------------------------------------------
__global__ void attn_scan_kernel(const float* __restrict__ YR,
                                 const float* __restrict__ YK,
                                 const float* __restrict__ YV,
                                 h16* __restrict__ RH) {
  int i = blockIdx.x * 256 + threadIdx.x;          // over N_*D_
  if (i >= N_ * D_) return;
  int d = i & (D_ - 1), n = i >> 8;
  float vr = 0.f, vk = 0.f, vv = 0.f, h = 0.f;
#pragma unroll
  for (int t = 0; t < T_; ++t) {
    size_t a = ((size_t)n * T_ + t) * D_ + d;
    vr = 0.5f * (vr + YR[a]); float r = (vr >= 1.0f) ? 1.0f : 0.0f; vr *= (1.0f - r);
    vk = 0.5f * (vk + YK[a]); float k = (vk >= 1.0f) ? 1.0f : 0.0f; vk *= (1.0f - k);
    vv = 0.5f * (vv + YV[a]); float v = (vv >= 1.0f) ? 1.0f : 0.0f; vv *= (1.0f - v);
    h = h * 0.9f + k * v;
    RH[a] = (h16)(r * h);
  }
}

// ---------------------------------------------------------------------------
// LayerNorm over D=256 per row (one wave32 per row), optional residual add.
// ---------------------------------------------------------------------------
__global__ void ln_kernel(const float* __restrict__ A, const float* __restrict__ R,
                          const float* __restrict__ g, const float* __restrict__ be,
                          float* __restrict__ OF, h16* __restrict__ OH) {
  int row  = blockIdx.x * 8 + (threadIdx.x >> 5);
  int lane = threadIdx.x & 31;
  if (row >= NT_) return;
  size_t base = (size_t)row * D_;
  float x[8]; float sum = 0.f;
#pragma unroll
  for (int j = 0; j < 8; ++j) {
    int d = lane + j * 32;
    float v = A[base + d] + (R ? R[base + d] : 0.0f);
    x[j] = v; sum += v;
  }
#pragma unroll
  for (int m = 16; m >= 1; m >>= 1) sum += __shfl_xor(sum, m, 32);
  float mean = sum * (1.0f / D_);
  float sq = 0.f;
#pragma unroll
  for (int j = 0; j < 8; ++j) { float dd = x[j] - mean; sq += dd * dd; }
#pragma unroll
  for (int m = 16; m >= 1; m >>= 1) sq += __shfl_xor(sq, m, 32);
  float rstd = rsqrtf(sq * (1.0f / D_) + 1e-5f);
#pragma unroll
  for (int j = 0; j < 8; ++j) {
    int d = lane + j * 32;
    float o = (x[j] - mean) * rstd * g[d] + be[d];
    OF[base + d] = o;
    OH[base + d] = (h16)o;
  }
}

// ---------------------------------------------------------------------------
// FFN LIF1: s1 = LIF(G) over t, emit f16 spikes
// ---------------------------------------------------------------------------
__global__ void lif1_kernel(const float* __restrict__ G, h16* __restrict__ S1H) {
  int i = blockIdx.x * 256 + threadIdx.x;          // over N_*F_
  if (i >= N_ * F_) return;
  int f = i & (F_ - 1), n = i >> 10;
  float v1 = 0.f;
#pragma unroll
  for (int t = 0; t < T_; ++t) {
    size_t a = ((size_t)n * T_ + t) * F_ + f;
    v1 = 0.5f * (v1 + G[a]);
    float s = (v1 >= 1.0f) ? 1.0f : 0.0f; v1 *= (1.0f - s);
    S1H[a] = (h16)s;
  }
}

// ---------------------------------------------------------------------------
// FFN LIF2 + residual: X2 = X1 + LIF(HF)
// ---------------------------------------------------------------------------
__global__ void lif2res_kernel(const float* __restrict__ HF,
                               const float* __restrict__ X1F,
                               float* __restrict__ X2F) {
  int i = blockIdx.x * 256 + threadIdx.x;          // over N_*D_
  if (i >= N_ * D_) return;
  int d = i & (D_ - 1), n = i >> 8;
  float v2 = 0.f;
#pragma unroll
  for (int t = 0; t < T_; ++t) {
    size_t a = ((size_t)n * T_ + t) * D_ + d;
    v2 = 0.5f * (v2 + HF[a]);
    float s = (v2 >= 1.0f) ? 1.0f : 0.0f; v2 *= (1.0f - s);
    X2F[a] = X1F[a] + s;
  }
}

// ---------------------------------------------------------------------------
// Fused head: per (16-row, 64-vocab) tile, loop t: WMMA logits -> LIF ->
// accumulate spikes. Membrane state lives in registers; logits never stored.
// ---------------------------------------------------------------------------
__global__ void __launch_bounds__(128)
head_kernel(const h16* __restrict__ XH, const h16* __restrict__ WoutT,
            const float* __restrict__ bout, float* __restrict__ out) {
  __shared__ __align__(16) h16 sA[16 * D_];   //  8 KB: x rows at current t
  __shared__ __align__(16) h16 sB[64 * D_];   // 32 KB: Wout^T tile (stays resident)
  const int n0 = blockIdx.x * 16, v0 = blockIdx.y * 64;
  const int tid = threadIdx.x;
  const int w = tid >> 5, lane = tid & 31;
  const int colc = lane & 15, hh = lane >> 4;

  // Load B tile: 64 vocab cols x 256 K (16 v8h chunks per thread)
#pragma unroll
  for (int j = 0; j < 16; ++j) {
    int ch = tid + j * 128;                 // 2048 chunks total
    int c  = ch >> 5, cg = (ch & 31) * 8;
    *(v8h*)&sB[c * D_ + cg] = *(const v8h*)&WoutT[(size_t)(v0 + c) * D_ + cg];
  }
  const int col = v0 + w * 16 + colc;
  const float bi = bout[col];

  float vst[8], acc[8];
#pragma unroll
  for (int r = 0; r < 8; ++r) { vst[r] = 0.f; acc[r] = 0.f; }

  for (int t = 0; t < T_; ++t) {
    __syncthreads();
#pragma unroll
    for (int j = 0; j < 4; ++j) {           // 512 chunks of A
      int ch = tid + j * 128;
      int r  = ch >> 5, cg = (ch & 31) * 8;
      *(v8h*)&sA[r * D_ + cg] =
          *(const v8h*)&XH[(((size_t)(n0 + r)) * T_ + t) * D_ + cg];
    }
    __syncthreads();
    v8f c = {};
#pragma unroll
    for (int kk = 0; kk < 8; ++kk) {
      int k0 = kk * 32;
      FragH a, b;
      a.h[0] = *(const v8h*)&sA[colc * D_ + k0 + hh * 8];
      a.h[1] = *(const v8h*)&sA[colc * D_ + k0 + 16 + hh * 8];
      b.h[0] = *(const v8h*)&sB[(w * 16 + colc) * D_ + k0 + hh * 8];
      b.h[1] = *(const v8h*)&sB[(w * 16 + colc) * D_ + k0 + 16 + hh * 8];
      c = WMMA16(a.v, b.v, c);
    }
#pragma unroll
    for (int r = 0; r < 8; ++r) {           // head LIF, state in registers
      float lg = c[r] + bi;
      vst[r] = 0.5f * (vst[r] + lg);
      float s = (vst[r] >= 1.0f) ? 1.0f : 0.0f;
      vst[r] *= (1.0f - s);
      acc[r] += s;
    }
  }
#pragma unroll
  for (int r = 0; r < 8; ++r)
    out[(size_t)(n0 + r + 8 * hh) * V_ + col] = acc[r];
}

// ---------------------------------------------------------------------------
// Host launch
// ---------------------------------------------------------------------------
extern "C" void kernel_launch(void* const* d_in, const int* in_sizes, int n_in,
                              void* d_out, int out_size, void* d_ws, size_t ws_size,
                              hipStream_t stream) {
  const int*   ids   = (const int*)  d_in[0];
  const float* tok   = (const float*)d_in[1];
  const float* pos   = (const float*)d_in[2];
  const float* noise = (const float*)d_in[3];
  const float* unif  = (const float*)d_in[4];
  const float* Wr    = (const float*)d_in[5];
  const float* Wk    = (const float*)d_in[6];
  const float* Wv    = (const float*)d_in[7];
  const float* Wo    = (const float*)d_in[8];
  const float* W1    = (const float*)d_in[9];
  const float* b1    = (const float*)d_in[10];
  const float* W2    = (const float*)d_in[11];
  const float* b2    = (const float*)d_in[12];
  const float* g1    = (const float*)d_in[13];
  const float* be1   = (const float*)d_in[14];
  const float* g2    = (const float*)d_in[15];
  const float* be2   = (const float*)d_in[16];
  const float* Wout  = (const float*)d_in[17];
  const float* bout  = (const float*)d_in[18];
  float* out = (float*)d_out;

  size_t off = 0;
  auto alloc = [&](size_t bytes) -> void* {
    void* p = (char*)d_ws + off;
    off = (off + bytes + 255) & ~(size_t)255;
    return p;
  };
  h16*   XH   = (h16*)  alloc((size_t)NT_ * D_ * 2);
  float* XF   = (float*)alloc((size_t)NT_ * D_ * 4);
  float* YR   = (float*)alloc((size_t)NT_ * D_ * 4);
  float* YK   = (float*)alloc((size_t)NT_ * D_ * 4);
  float* YV   = (float*)alloc((size_t)NT_ * D_ * 4);
  h16*   RH   = (h16*)  alloc((size_t)NT_ * D_ * 2);
  float* ATT  = (float*)alloc((size_t)NT_ * D_ * 4);
  float* X1F  = (float*)alloc((size_t)NT_ * D_ * 4);
  h16*   X1H  = (h16*)  alloc((size_t)NT_ * D_ * 2);
  float* G    = (float*)alloc((size_t)NT_ * F_ * 4);
  h16*   S1H  = (h16*)  alloc((size_t)NT_ * F_ * 2);
  float* HF   = (float*)alloc((size_t)NT_ * D_ * 4);
  float* X2F  = (float*)alloc((size_t)NT_ * D_ * 4);
  h16*   WrT  = (h16*)  alloc((size_t)L_ * D_ * D_ * 2);
  h16*   WkT  = (h16*)  alloc((size_t)L_ * D_ * D_ * 2);
  h16*   WvT  = (h16*)  alloc((size_t)L_ * D_ * D_ * 2);
  h16*   WoT  = (h16*)  alloc((size_t)L_ * D_ * D_ * 2);
  h16*   W1T  = (h16*)  alloc((size_t)L_ * D_ * F_ * 2);
  h16*   W2T  = (h16*)  alloc((size_t)L_ * D_ * F_ * 2);
  h16*   WouT = (h16*)  alloc((size_t)D_ * V_ * 2);
  (void)ws_size; (void)in_sizes; (void)n_in; (void)out_size;

  // --- weight convert/transpose (f32 -> f16, N-major) ---
  const int DD = D_ * D_, DF = D_ * F_;
  for (int l = 0; l < L_; ++l) {
    convT_kernel<<<(DD + 255) / 256, 256, 0, stream>>>(Wr + (size_t)l * DD, WrT + (size_t)l * DD, D_, D_);
    convT_kernel<<<(DD + 255) / 256, 256, 0, stream>>>(Wk + (size_t)l * DD, WkT + (size_t)l * DD, D_, D_);
    convT_kernel<<<(DD + 255) / 256, 256, 0, stream>>>(Wv + (size_t)l * DD, WvT + (size_t)l * DD, D_, D_);
    convT_kernel<<<(DD + 255) / 256, 256, 0, stream>>>(Wo + (size_t)l * DD, WoT + (size_t)l * DD, D_, D_);
    convT_kernel<<<(DF + 255) / 256, 256, 0, stream>>>(W1 + (size_t)l * DF, W1T + (size_t)l * DF, D_, F_);
    convT_kernel<<<(DF + 255) / 256, 256, 0, stream>>>(W2 + (size_t)l * DF, W2T + (size_t)l * DF, F_, D_);
  }
  convT_kernel<<<((size_t)D_ * V_ + 255) / 256, 256, 0, stream>>>(Wout, WouT, D_, V_);

  // --- spike encoding ---
  encode_kernel<<<(N_ * D_) / 256, 256, 0, stream>>>(ids, tok, pos, noise, unif, XF, XH);

  // --- transformer blocks ---
  for (int l = 0; l < L_; ++l) {
    dim3 gdd(NT_ / 128, D_ / 64);
    gemm_wmma<<<gdd, 256, 0, stream>>>(XH, WrT + (size_t)l * DD, nullptr, YR, NT_, D_, D_);
    gemm_wmma<<<gdd, 256, 0, stream>>>(XH, WkT + (size_t)l * DD, nullptr, YK, NT_, D_, D_);
    gemm_wmma<<<gdd, 256, 0, stream>>>(XH, WvT + (size_t)l * DD, nullptr, YV, NT_, D_, D_);
    attn_scan_kernel<<<(N_ * D_) / 256, 256, 0, stream>>>(YR, YK, YV, RH);
    gemm_wmma<<<gdd, 256, 0, stream>>>(RH, WoT + (size_t)l * DD, nullptr, ATT, NT_, D_, D_);
    ln_kernel<<<NT_ / 8, 256, 0, stream>>>(XF, ATT, g1 + l * D_, be1 + l * D_, X1F, X1H);
    dim3 gdf(NT_ / 128, F_ / 64);
    gemm_wmma<<<gdf, 256, 0, stream>>>(X1H, W1T + (size_t)l * DF, b1 + l * F_, G, NT_, F_, D_);
    lif1_kernel<<<(N_ * F_) / 256, 256, 0, stream>>>(G, S1H);
    gemm_wmma<<<gdd, 256, 0, stream>>>(S1H, W2T + (size_t)l * DF, b2 + l * D_, HF, NT_, D_, F_);
    lif2res_kernel<<<(N_ * D_) / 256, 256, 0, stream>>>(HF, X1F, X2F);
    ln_kernel<<<NT_ / 8, 256, 0, stream>>>(X2F, nullptr, g2 + l * D_, be2 + l * D_, XF, XH);
  }

  // --- fused head GEMM + head LIF + spike accumulation ---
  head_kernel<<<dim3(N_ / 16, V_ / 64), 128, 0, stream>>>(XH, WouT, bout, out);
}
